// MotifEncoder_43258910605922
// MI455X (gfx1250) — matrix operand order repaired and checked by reading, not compiled, not run on hardware
//
#include <hip/hip_runtime.h>

// ---------------------------------------------------------------------------
// GINE GNN (2 conv layers + sum-pool + FC) for MI455X / gfx1250.
// Edge phase: fused edge-linear + gather + relu + atomic scatter (mem-bound).
// Node MLPs / FC: f16 WMMA (v_wmma_f32_16x16x32_f16) with f32 accumulate,
// weights staged in LDS pre-swizzled to B-fragment layout.
// ---------------------------------------------------------------------------

typedef __attribute__((ext_vector_type(16))) _Float16 v16h;
typedef __attribute__((ext_vector_type(8)))  float    v8f;

#define N_NODES 50000
#define N_EDGES 800000
#define N_GRAPHS 512

// ---------------------------------------------------------------- zero ----
__global__ void zero_kernel(float* __restrict__ p, int n) {
  int i = blockIdx.x * blockDim.x + threadIdx.x;
  if (i < n) p[i] = 0.0f;
}

// ------------------------------------------------------------ edge msg ----
// One thread per (edge, 4 output features):
//   msg = relu(x[src] + edge_attr@We + be); atomicAdd into agg[dst].
template <int D>
__global__ void edge_msg_kernel(const float* __restrict__ X,    // [N, D]
                                const float* __restrict__ EA,   // [E, 16]
                                const int*   __restrict__ SRC,
                                const int*   __restrict__ DST,
                                const float* __restrict__ We,   // [16, D]
                                const float* __restrict__ be,   // [D]
                                float*       __restrict__ AGG,  // [N, D]
                                int nEdges) {
  constexpr int QPE = D / 4;
  int tid = blockIdx.x * blockDim.x + threadIdx.x;
  if (tid >= nEdges * QPE) return;
  int e = tid / QPE;
  int j = (tid - e * QPE) * 4;

  int s  = SRC[e];
  int dd = DST[e];
  const float* ea = EA + e * 16;

  float a0 = be[j + 0], a1 = be[j + 1], a2 = be[j + 2], a3 = be[j + 3];
#pragma unroll
  for (int k = 0; k < 16; ++k) {
    float ek = ea[k];
    const float* w = We + k * D + j;
    a0 = fmaf(ek, w[0], a0);
    a1 = fmaf(ek, w[1], a1);
    a2 = fmaf(ek, w[2], a2);
    a3 = fmaf(ek, w[3], a3);
  }
  const float4 xs = *(const float4*)(X + s * D + j);
  a0 = fmaxf(a0 + xs.x, 0.0f);
  a1 = fmaxf(a1 + xs.y, 0.0f);
  a2 = fmaxf(a2 + xs.z, 0.0f);
  a3 = fmaxf(a3 + xs.w, 0.0f);

  float* ag = AGG + dd * D + j;
  unsafeAtomicAdd(ag + 0, a0);
  unsafeAtomicAdd(ag + 1, a1);
  unsafeAtomicAdd(ag + 2, a2);
  unsafeAtomicAdd(ag + 3, a3);
}

// ------------------------------------------------------------ WMMA GEMM ---
// OUT[M,128] = act( (A (+A2)) [M,K] @ W [K,128] + bias ),  K in {96,128}.
// One wave per 16-row M tile; 8 N-tiles of 16 each; K stepped by 32.
// W staged once per workgroup into LDS in exact B-fragment order so each
// B tile is one aligned 32B LDS read per lane. HAS_A2 templated so the
// fused add has no exec-mask branch in the inner loop.
#define GEMM_WAVES 8
template <bool HAS_A2>
__global__ void gemm128_kernel(const float* __restrict__ A,
                               const float* __restrict__ A2,
                               const float* __restrict__ W,    // [K,128]
                               const float* __restrict__ bias, // [128]
                               float*       __restrict__ OUT,  // [M,128]
                               int M, int K, int doRelu) {
  __shared__ _Float16 ldsW[4 * 8 * 32 * 16];  // [kk][j][lane][e], 32 KB max

  const int tid = threadIdx.x;
  const int nk  = K >> 5;  // K/32

  // Stage W -> LDS, swizzled to B layout: B[k][n], lane l holds
  // k = kk*32 + (l>>4)*8 + e (e<8) / kk*32 + 16 + (l>>4)*8 + e-8, n = j*16 + (l&15).
  for (int idx = tid; idx < nk * 4096; idx += blockDim.x) {
    int e  = idx & 15;
    int l  = (idx >> 4) & 31;
    int j  = (idx >> 9) & 7;
    int kk = idx >> 12;
    int krow = kk * 32 + ((e >> 3) << 4) + ((l >> 4) << 3) + (e & 7);
    int col  = j * 16 + (l & 15);
    ldsW[idx] = (_Float16)W[krow * 128 + col];
  }
  __syncthreads();

  const int wave = tid >> 5;
  const int lane = tid & 31;
  const int mt   = blockIdx.x * GEMM_WAVES + wave;
  if (mt * 16 >= M) return;          // wave-uniform: EXEC stays all-1s

  const int h  = lane >> 4;   // half-wave
  const int mr = lane & 15;   // A row within tile
  const int nD = lane & 15;   // C/D column within tile

  const float* arow  = A + (mt * 16 + mr) * K;
  const float* arow2 = HAS_A2 ? (A2 + (mt * 16 + mr) * K) : nullptr;

  v8f acc[8];
#pragma unroll
  for (int j = 0; j < 8; ++j) {
    float bv = bias[j * 16 + nD];
#pragma unroll
    for (int r = 0; r < 8; ++r) acc[j][r] = bv;
  }

  for (int kk = 0; kk < nk; ++kk) {
    const int k0 = kk * 32;
    // A fragment: lane half h holds K = k0+h*8+e (e<8) and K = k0+16+h*8+e.
    const float4 x0 = *(const float4*)(arow + k0 + h * 8);
    const float4 x1 = *(const float4*)(arow + k0 + h * 8 + 4);
    const float4 y0 = *(const float4*)(arow + k0 + 16 + h * 8);
    const float4 y1 = *(const float4*)(arow + k0 + 16 + h * 8 + 4);
    float lo[8] = {x0.x, x0.y, x0.z, x0.w, x1.x, x1.y, x1.z, x1.w};
    float hi[8] = {y0.x, y0.y, y0.z, y0.w, y1.x, y1.y, y1.z, y1.w};
    if (HAS_A2) {
      const float4 u0 = *(const float4*)(arow2 + k0 + h * 8);
      const float4 u1 = *(const float4*)(arow2 + k0 + h * 8 + 4);
      const float4 w0 = *(const float4*)(arow2 + k0 + 16 + h * 8);
      const float4 w1 = *(const float4*)(arow2 + k0 + 16 + h * 8 + 4);
      lo[0] += u0.x; lo[1] += u0.y; lo[2] += u0.z; lo[3] += u0.w;
      lo[4] += u1.x; lo[5] += u1.y; lo[6] += u1.z; lo[7] += u1.w;
      hi[0] += w0.x; hi[1] += w0.y; hi[2] += w0.z; hi[3] += w0.w;
      hi[4] += w1.x; hi[5] += w1.y; hi[6] += w1.z; hi[7] += w1.w;
    }
    v16h av;
#pragma unroll
    for (int e = 0; e < 8; ++e) {
      av[e]     = (_Float16)lo[e];
      av[8 + e] = (_Float16)hi[e];
    }
#pragma unroll
    for (int j = 0; j < 8; ++j) {
      const v16h bvv = *(const v16h*)(ldsW + ((((kk * 8 + j) * 32) + lane) << 4));
      acc[j] = __builtin_amdgcn_wmma_f32_16x16x32_f16(
          false, av, false, bvv, (short)0, acc[j], false, false);
    }
  }

  // Store D: element r of acc[j] -> row mt*16 + h*8 + r, col j*16 + nD.
#pragma unroll
  for (int j = 0; j < 8; ++j)
#pragma unroll
    for (int r = 0; r < 8; ++r) {
      float v = acc[j][r];
      if (doRelu) v = fmaxf(v, 0.0f);
      OUT[(mt * 16 + h * 8 + r) * 128 + j * 16 + nD] = v;
    }
}

// ----------------------------------------------------------------- pool ---
__global__ void pool_kernel(const float* __restrict__ H,  // [N,128]
                            const int*   __restrict__ B,  // [N]
                            float*       __restrict__ P,  // [G,128]
                            int nNodes) {
  int tid = blockIdx.x * blockDim.x + threadIdx.x;
  if (tid >= nNodes * 32) return;
  int n = tid >> 5;
  int j = (tid & 31) * 4;
  int g = B[n];
  const float4 hv = *(const float4*)(H + n * 128 + j);
  float* pv = P + g * 128 + j;
  unsafeAtomicAdd(pv + 0, hv.x);
  unsafeAtomicAdd(pv + 1, hv.y);
  unsafeAtomicAdd(pv + 2, hv.z);
  unsafeAtomicAdd(pv + 3, hv.w);
}

// --------------------------------------------------------------- launch ---
extern "C" void kernel_launch(void* const* d_in, const int* in_sizes, int n_in,
                              void* d_out, int out_size, void* d_ws, size_t ws_size,
                              hipStream_t stream) {
  const float* x    = (const float*)d_in[0];
  const int*   ei   = (const int*)d_in[1];
  const float* ea   = (const float*)d_in[2];
  const int*   bat  = (const int*)d_in[3];
  const float* We1  = (const float*)d_in[4];
  const float* be1  = (const float*)d_in[5];
  const float* W1a  = (const float*)d_in[6];
  const float* b1a  = (const float*)d_in[7];
  const float* W1b  = (const float*)d_in[8];
  const float* b1b  = (const float*)d_in[9];
  const float* We2  = (const float*)d_in[10];
  const float* be2  = (const float*)d_in[11];
  const float* W2a  = (const float*)d_in[12];
  const float* b2a  = (const float*)d_in[13];
  const float* W2b  = (const float*)d_in[14];
  const float* b2b  = (const float*)d_in[15];
  const float* Wfc  = (const float*)d_in[16];
  const float* bfc  = (const float*)d_in[17];

  const int* src = ei;
  const int* dst = ei + N_EDGES;

  // Workspace layout (floats): agg1 | agg2 | pooled | tbuf | h1 | h2o
  float* ws     = (float*)d_ws;
  float* agg1   = ws;                               // 50000*96
  float* agg2   = agg1 + N_NODES * 96;              // 50000*128
  float* pooled = agg2 + N_NODES * 128;             // 512*128
  float* tbuf   = pooled + N_GRAPHS * 128;          // 50000*128
  float* h1     = tbuf + N_NODES * 128;             // 50000*128
  float* h2o    = h1 + N_NODES * 128;               // 50000*128

  const int nZero = N_NODES * 96 + N_NODES * 128 + N_GRAPHS * 128;
  zero_kernel<<<(nZero + 255) / 256, 256, 0, stream>>>(ws, nZero);

  // Layer 1: msg/scatter (d=96), then MLP 96->128->128 (both ReLU).
  {
    int nth = N_EDGES * (96 / 4);
    edge_msg_kernel<96><<<(nth + 255) / 256, 256, 0, stream>>>(
        x, ea, src, dst, We1, be1, agg1, N_EDGES);
  }
  {
    int blocks = (N_NODES / 16 + GEMM_WAVES - 1) / GEMM_WAVES;  // 3125 tiles
    gemm128_kernel<true><<<blocks, 32 * GEMM_WAVES, 0, stream>>>(
        x, agg1, W1a, b1a, tbuf, N_NODES, 96, 1);
    gemm128_kernel<false><<<blocks, 32 * GEMM_WAVES, 0, stream>>>(
        tbuf, nullptr, W1b, b1b, h1, N_NODES, 128, 1);
  }

  // Layer 2: msg/scatter (d=128), then MLP 128->128->128 (both ReLU).
  {
    int nth = N_EDGES * (128 / 4);
    edge_msg_kernel<128><<<(nth + 255) / 256, 256, 0, stream>>>(
        h1, ea, src, dst, We2, be2, agg2, N_EDGES);
  }
  {
    int blocks = (N_NODES / 16 + GEMM_WAVES - 1) / GEMM_WAVES;
    gemm128_kernel<true><<<blocks, 32 * GEMM_WAVES, 0, stream>>>(
        h1, agg2, W2a, b2a, tbuf, N_NODES, 128, 1);
    gemm128_kernel<false><<<blocks, 32 * GEMM_WAVES, 0, stream>>>(
        tbuf, nullptr, W2b, b2b, h2o, N_NODES, 128, 1);
  }

  // Pool + final FC (no ReLU) -> d_out [512,128].
  {
    int nth = N_NODES * 32;
    pool_kernel<<<(nth + 255) / 256, 256, 0, stream>>>(h2o, bat, pooled, N_NODES);
  }
  {
    int blocks = (N_GRAPHS / 16 + GEMM_WAVES - 1) / GEMM_WAVES;  // 32 tiles
    gemm128_kernel<false><<<blocks, 32 * GEMM_WAVES, 0, stream>>>(
        pooled, nullptr, Wfc, bfc, (float*)d_out, N_GRAPHS, 128, 0);
  }
}
